// InfomationLayer_66786741453338
// MI455X (gfx1250) — matrix operand order repaired and checked by reading, not compiled, not run on hardware
//
#include <hip/hip_runtime.h>

typedef __attribute__((ext_vector_type(16))) __bf16 v16bf;
typedef __attribute__((ext_vector_type(8)))  __bf16 bf16x8;
typedef __attribute__((ext_vector_type(8)))  float  v8f;
typedef int v4i __attribute__((vector_size(16)));

#define GLOBAL_AS __attribute__((address_space(1)))
#define LDS_AS    __attribute__((address_space(3)))

#define WMMA_BF16(a, b, c) \
  __builtin_amdgcn_wmma_f32_16x16x32_bf16(false, (a), false, (b), (short)0, (c), false, false)

#if __has_builtin(__builtin_amdgcn_global_load_async_to_lds_b128) && \
    __has_builtin(__builtin_amdgcn_s_wait_asynccnt)
#define USE_ASYNC_LDS 1
#else
#define USE_ASYNC_LDS 0
#endif

static __device__ inline v16bf concat8(bf16x8 lo, bf16x8 hi) {
  return __builtin_shufflevector(lo, hi, 0,1,2,3,4,5,6,7,8,9,10,11,12,13,14,15);
}

// A fragment 16x32 bf16 (row-major source, row stride in elements).
// ISA layout: lanes 0-15 -> M=lane, VGPR0..3 = K 0..7, VGPR4..7 = K 16..23;
//             lanes 16-31 -> M=lane-16, K 8..15 and K 24..31.
static __device__ inline v16bf load_a(const __bf16* base, int stride, int lane) {
  int m  = lane & 15;
  int hi = (lane >> 4) << 3;                    // 0 or 8
  const __bf16* row = base + (size_t)m * stride + hi;
  bf16x8 c0 = *(const bf16x8*)(row);            // K = hi .. hi+7
  bf16x8 c1 = *(const bf16x8*)(row + 16);       // K = 16+hi .. 16+hi+7
  return concat8(c0, c1);
}

// B fragment 32x16 bf16: lane l supplies column n=l%16, K = hi..hi+7 and 16+hi..16+hi+7.
static __device__ inline v16bf load_b(const __bf16* base, int stride, int lane) {
  int n  = lane & 15;
  int hi = (lane >> 4) << 3;
  v16bf r;
#pragma unroll
  for (int s = 0; s < 8; ++s) {
    r[s]     = base[(size_t)(hi + s) * stride + n];
    r[s + 8] = base[(size_t)(16 + hi + s) * stride + n];
  }
  return r;
}

#if USE_ASYNC_LDS
// Stage a 32x512 bf16 V tile (32 KB) from global into LDS with async copies.
// 2048 16-byte chunks, 128 threads -> 16 async ops per thread (ASYNCcnt += 16).
static __device__ inline void stage_v_async(const __bf16* gsrc, __bf16* ldst, int tid) {
#pragma unroll
  for (int i = 0; i < 16; ++i) {
    int c   = tid + i * 128;              // chunk id 0..2047
    int row = c >> 6;                     // 64 chunks (of 8 elems) per 512-elem row
    int col = (c & 63) * 8;
    __builtin_amdgcn_global_load_async_to_lds_b128(
        (GLOBAL_AS v4i*)(gsrc + (size_t)row * 512 + col),
        (LDS_AS v4i*)(ldst + row * 512 + col),
        0, 0);
  }
}
#endif

// ---------------- Pass 0: dtype conversion / layout prep ----------------

__global__ void cvt_xy_kernel(const float* __restrict__ x, const float* __restrict__ y,
                              __bf16* __restrict__ xybf) {
  size_t idx = (size_t)blockIdx.x * 256 + threadIdx.x;   // 0 .. 16384*512-1
  int    d    = (int)(idx & 511);
  size_t srow = idx >> 9;                                 // b*2048 + s
  int    b    = (int)(srow >> 11);
  int    s    = (int)(srow & 2047);
  float v = (s < 1024) ? x[((size_t)b * 1024 + s) * 512 + d]
                       : y[((size_t)b * 1024 + (s - 1024)) * 512 + d];
  xybf[idx] = (__bf16)v;
}

__global__ void cvt_w_kernel(const float* __restrict__ Wq, const float* __restrict__ Wk,
                             const float* __restrict__ Wv, __bf16* __restrict__ wtq,
                             __bf16* __restrict__ wtk, __bf16* __restrict__ wtv) {
  int idx = blockIdx.x * 256 + threadIdx.x;   // out index = d*512 + e
  int d = idx >> 9, e = idx & 511;
  size_t src = (size_t)e * 512 + d;           // W[e][d]
  wtq[idx] = (__bf16)Wq[src];
  wtk[idx] = (__bf16)Wk[src];
  wtv[idx] = (__bf16)Wv[src];
}

// ---------------- Pass 1: fused Q/K/V projection GEMM ----------------
// grid (1024, 8), block 128 (4 waves). Wave computes one 16x16 tile for Q, K, V.
// Q, V stored row-major [16384,512]; K stored transposed Kt[b][e][s].

__global__ void qkv_gemm_kernel(const __bf16* __restrict__ xybf,
                                const __bf16* __restrict__ wtq,
                                const __bf16* __restrict__ wtk,
                                const __bf16* __restrict__ wtv,
                                __bf16* __restrict__ Qbf,
                                __bf16* __restrict__ Kt,
                                __bf16* __restrict__ Vbf) {
  int lane = threadIdx.x & 31, wave = threadIdx.x >> 5;
  int row0 = blockIdx.x * 16;                 // global token row
  int e0   = blockIdx.y * 64 + wave * 16;     // output column tile
  v8f aq = {}, ak = {}, av = {};
#pragma unroll 4
  for (int kk = 0; kk < 512; kk += 32) {
    v16bf a = load_a(xybf + (size_t)row0 * 512 + kk, 512, lane);
    size_t brow = (size_t)kk * 512 + e0;
    aq = WMMA_BF16(a, load_b(wtq + brow, 512, lane), aq);
    ak = WMMA_BF16(a, load_b(wtk + brow, 512, lane), ak);
    av = WMMA_BF16(a, load_b(wtv + brow, 512, lane), av);
  }
  int hi = (lane >> 4) << 3, nn = lane & 15;
  int e = e0 + nn;
#pragma unroll
  for (int r = 0; r < 8; ++r) {
    int srow = row0 + r + hi;                 // C layout: M = r (+8 for hi lanes)
    Qbf[(size_t)srow * 512 + e] = (__bf16)aq[r];
    Vbf[(size_t)srow * 512 + e] = (__bf16)av[r];
    int b = srow >> 11, sl = srow & 2047;
    Kt[((size_t)b * 512 + e) * 2048 + sl] = (__bf16)ak[r];
  }
}

// ---------------- Pass 2: fused attention (scores -> softmax -> P@V) ----------------
// grid 1024, block 128 (4 waves). Workgroup owns 16 query rows.
// LDS: [red 16*8 f32][scores 16*2048 f32][P 16*2048 bf16][Vtile 2x32x512 bf16]
//      = 512 + 131072 + 65536 + 65536 = 262656 B (< 320 KB/WGP)

__global__ void attn_kernel(const __bf16* __restrict__ Qbf,
                            const __bf16* __restrict__ Kt,
                            const __bf16* __restrict__ Vbf,
                            float* __restrict__ out) {
  extern __shared__ __align__(16) char smem[];
  float*  red   = (float*)smem;                                  // 16*8
  float*  sc    = (float*)(smem + 512);                          // 16*2048 f32
  __bf16* P     = (__bf16*)(smem + 512 + 16 * 2048 * 4);         // 16*2048 bf16
  __bf16* vtile = (__bf16*)(smem + 512 + 16 * 2048 * 4 + 16 * 2048 * 2); // 2*32*512

  int tid = threadIdx.x, lane = tid & 31, wave = tid >> 5;
  int q0 = blockIdx.x * 16;                 // global query row
  int b  = q0 >> 11;
  int hi = (lane >> 4) << 3, nn = lane & 15;

  // ---- Stage 1: S[16, 2048] = Q @ K^T * scale ----
  v16bf Aq[16];
#pragma unroll
  for (int dk = 0; dk < 16; ++dk)
    Aq[dk] = load_a(Qbf + (size_t)q0 * 512 + dk * 32, 512, lane);

  const __bf16* KtB = Kt + (size_t)b * 512 * 2048;
  const float scale = 0.044194173824159216f;           // 1/sqrt(512)
  for (int t = 0; t < 32; ++t) {                       // wave covers 512 key cols
    int n0 = wave * 512 + t * 16;
    v8f acc = {};
#pragma unroll
    for (int dk = 0; dk < 16; ++dk)
      acc = WMMA_BF16(Aq[dk], load_b(KtB + (size_t)(dk * 32) * 2048 + n0, 2048, lane), acc);
#pragma unroll
    for (int r = 0; r < 8; ++r)
      sc[(r + hi) * 2048 + n0 + nn] = acc[r] * scale;
  }
  __syncthreads();

  // ---- Softmax over each of the 16 rows (8 threads per row, 256 elems each) ----
  int row = tid >> 3, sub = tid & 7;
  float* rp = sc + row * 2048 + sub * 256;
  float mx = -3.4e38f;
  for (int j = 0; j < 256; ++j) mx = fmaxf(mx, rp[j]);
  red[row * 8 + sub] = mx;
  __syncthreads();
  float rmx = -3.4e38f;
  for (int k = 0; k < 8; ++k) rmx = fmaxf(rmx, red[row * 8 + k]);
  __syncthreads();
  float sm = 0.f;
  for (int j = 0; j < 256; ++j) { float e = __expf(rp[j] - rmx); rp[j] = e; sm += e; }
  red[row * 8 + sub] = sm;
  __syncthreads();
  float rsum = 0.f;
  for (int k = 0; k < 8; ++k) rsum += red[row * 8 + k];
  float inv = 1.0f / rsum;
  __bf16* pp = P + row * 2048 + sub * 256;
  for (int j = 0; j < 256; ++j) pp[j] = (__bf16)(rp[j] * inv);
  __syncthreads();

  // ---- Stage 2: out[16, 512] = P @ V ; wave owns 128 output columns ----
  const __bf16* Vb = Vbf + (size_t)b * 2048 * 512;
  int e0 = wave * 128;
  v8f acc[8] = {};

#if USE_ASYNC_LDS
  // Double-buffered async staging of 32x512 V tiles through LDS.
  __bf16* vt0 = vtile;
  __bf16* vt1 = vtile + 32 * 512;
  stage_v_async(Vb, vt0, tid);                       // ASYNCcnt = 16
  for (int kt = 0; kt < 64; ++kt) {
    __bf16* cur = (kt & 1) ? vt1 : vt0;
    __bf16* nxt = (kt & 1) ? vt0 : vt1;
    if (kt + 1 < 64) {
      stage_v_async(Vb + (size_t)(kt + 1) * 32 * 512, nxt, tid);  // overlap next tile
      __builtin_amdgcn_s_wait_asynccnt(16);          // current tile's 16 copies done
    } else {
      __builtin_amdgcn_s_wait_asynccnt(0);
    }
    __syncthreads();                                 // all waves' copies visible
    v16bf a = load_a(P + kt * 32, 2048, lane);       // ds_load_b128 from LDS P
    const __bf16* vrow = cur + e0;
#pragma unroll
    for (int j = 0; j < 8; ++j)
      acc[j] = WMMA_BF16(a, load_b(vrow + j * 16, 512, lane), acc[j]);
    __syncthreads();                                 // reads done before tile reuse
  }
#else
  (void)vtile;
  for (int k0 = 0; k0 < 2048; k0 += 32) {
    v16bf a = load_a(P + k0, 2048, lane);
    const __bf16* vrow = Vb + (size_t)k0 * 512 + e0;
#pragma unroll
    for (int j = 0; j < 8; ++j)
      acc[j] = WMMA_BF16(a, load_b(vrow + j * 16, 512, lane), acc[j]);
  }
#endif

#pragma unroll
  for (int j = 0; j < 8; ++j)
#pragma unroll
    for (int r = 0; r < 8; ++r)
      out[(size_t)(q0 + r + hi) * 512 + e0 + j * 16 + nn] = acc[j][r];
}

// ---------------- Host launcher ----------------

extern "C" void kernel_launch(void* const* d_in, const int* in_sizes, int n_in,
                              void* d_out, int out_size, void* d_ws, size_t ws_size,
                              hipStream_t stream) {
  const float* x  = (const float*)d_in[0];
  const float* y  = (const float*)d_in[1];
  const float* Wq = (const float*)d_in[2];
  const float* Wk = (const float*)d_in[3];
  const float* Wv = (const float*)d_in[4];
  float* out = (float*)d_out;

  // Workspace layout (bf16 elements): xybf | wtq wtk wtv | Qbf | Kt | Vbf (~68.7 MB)
  __bf16* ws   = (__bf16*)d_ws;
  __bf16* xybf = ws;
  __bf16* wtq  = xybf + (size_t)16384 * 512;
  __bf16* wtk  = wtq + 512 * 512;
  __bf16* wtv  = wtk + 512 * 512;
  __bf16* Qbf  = wtv + 512 * 512;
  __bf16* Kt   = Qbf + (size_t)16384 * 512;
  __bf16* Vbf  = Kt  + (size_t)16384 * 512;

  // Pass 0: conversions / transposes
  cvt_xy_kernel<<<(16384 * 512) / 256, 256, 0, stream>>>(x, y, xybf);
  cvt_w_kernel<<<(512 * 512) / 256, 256, 0, stream>>>(Wq, Wk, Wv, wtq, wtk, wtv);

  // Pass 1: QKV projections
  qkv_gemm_kernel<<<dim3(1024, 8), 128, 0, stream>>>(xybf, wtq, wtk, wtv, Qbf, Kt, Vbf);

  // Pass 2: fused attention
  size_t smemBytes = 512 + (size_t)16 * 2048 * 4 + (size_t)16 * 2048 * 2
                   + (size_t)2 * 32 * 512 * 2;       // 262656 B
  attn_kernel<<<1024, 128, smemBytes, stream>>>(Qbf, Kt, Vbf, out);
}
